// TwoStageAttentionLayer_22969485099075
// MI455X (gfx1250) — compile-verified, matrix operand order checked by reading, not run on hardware
//
#include <hip/hip_runtime.h>
#include <hip/hip_bf16.h>

typedef __attribute__((ext_vector_type(16))) __bf16 v16bf;
typedef __attribute__((ext_vector_type(4)))  __bf16 v4bf;
typedef __attribute__((ext_vector_type(8)))  float  v8f;

// ---------------------------------------------------------------------------
// CDNA5 WMMA fragment index helpers (wave32, 16x16x32 bf16)
//  A (16x32): lane -> row M = lane%16, half = lane/16;
//             element e -> K = e + 8*half + (e>=8 ? 8 : 0)
//  B (32x16): lane -> col N = lane%16, same K mapping
//  C/D f32 (16x16): vgpr v -> M = v + 8*half, N = lane%16
// All LDS tiles are laid out so each lane's 16 fragment elements are two
// contiguous 8-element (16B) runs -> ds_load_b128 pairs.
// ---------------------------------------------------------------------------
__device__ __forceinline__ int kmap(int e, int half) {
  return e + half * 8 + ((e >= 8) ? 8 : 0);
}

__device__ __forceinline__ float gelu_exact(float x) {
  return 0.5f * x * (1.0f + erff(x * 0.70710678118654752f));
}

// ---------------------------------------------------------------------------
// GEMM: Y[M,N] = act(X[M,K] @ W[K,N] + bias[N])
// grid = (M/64, N/64), block = 256 threads (8 waves).
// Wave w: row tile (w&3)*16, col half (w>>2)*32 -> two 16x16 WMMA tiles.
// Double-buffered LDS; register-staged global loads; 1 barrier per K-step.
// As  : [row][k]  (A frag reads contiguous k)
// BsT : [col][k]  (B frag reads contiguous k)
// ---------------------------------------------------------------------------
__global__ __launch_bounds__(256) void gemm_bias_act(
    const float* __restrict__ X, const float* __restrict__ W,
    const float* __restrict__ bias, float* __restrict__ Y,
    int K, int N, int act) {
  __shared__ __bf16 As[2][64][40];
  __shared__ __bf16 BsT[2][64][40];

  const int tid  = threadIdx.x;
  const int wv   = tid >> 5, lane = tid & 31;
  const int rw   = wv & 3,  cw   = wv >> 2;
  const int m    = lane & 15, half = lane >> 4;
  const int m0   = blockIdx.x * 64, n0 = blockIdx.y * 64;

  // per-thread float4 staging slots (A: 64x32 tile, B: 32x64 tile)
  const int a_r = tid >> 3, a_c = (tid & 7) * 4;    // + row 32 for 2nd slot
  const int b_r = tid >> 4, b_c = (tid & 15) * 4;   // + row 16 for 2nd slot

  float4 xa0, xa1, xb0, xb1;
  v8f acc0 = {0.f,0.f,0.f,0.f,0.f,0.f,0.f,0.f};
  v8f acc1 = acc0;

  const int nk = K >> 5;

  auto gload = [&](int k0) {
    xa0 = *(const float4*)&X[(size_t)(m0 + a_r) * K + k0 + a_c];
    xa1 = *(const float4*)&X[(size_t)(m0 + a_r + 32) * K + k0 + a_c];
    xb0 = *(const float4*)&W[(size_t)(k0 + b_r) * N + n0 + b_c];
    xb1 = *(const float4*)&W[(size_t)(k0 + b_r + 16) * N + n0 + b_c];
  };
  auto lstore = [&](int buf) {
    v4bf pa0 = { (__bf16)xa0.x, (__bf16)xa0.y, (__bf16)xa0.z, (__bf16)xa0.w };
    v4bf pa1 = { (__bf16)xa1.x, (__bf16)xa1.y, (__bf16)xa1.z, (__bf16)xa1.w };
    *(v4bf*)&As[buf][a_r][a_c]      = pa0;
    *(v4bf*)&As[buf][a_r + 32][a_c] = pa1;
    BsT[buf][b_c + 0][b_r] = (__bf16)xb0.x;
    BsT[buf][b_c + 1][b_r] = (__bf16)xb0.y;
    BsT[buf][b_c + 2][b_r] = (__bf16)xb0.z;
    BsT[buf][b_c + 3][b_r] = (__bf16)xb0.w;
    BsT[buf][b_c + 0][b_r + 16] = (__bf16)xb1.x;
    BsT[buf][b_c + 1][b_r + 16] = (__bf16)xb1.y;
    BsT[buf][b_c + 2][b_r + 16] = (__bf16)xb1.z;
    BsT[buf][b_c + 3][b_r + 16] = (__bf16)xb1.w;
  };

  gload(0);
  lstore(0);
  __syncthreads();

  for (int kt = 0; kt < nk; ++kt) {
    const int cur = kt & 1;
    if (kt + 1 < nk) gload((kt + 1) << 5);

    const __bf16* arow  = &As[cur][rw * 16 + m][0];
    const __bf16* brow0 = &BsT[cur][cw * 32 + m][0];
    const __bf16* brow1 = &BsT[cur][cw * 32 + 16 + m][0];
    v16bf a, b0, b1;
#pragma unroll
    for (int e = 0; e < 16; ++e) {
      const int kk = kmap(e, half);
      a[e]  = arow[kk];
      b0[e] = brow0[kk];
      b1[e] = brow1[kk];
    }
    acc0 = __builtin_amdgcn_wmma_f32_16x16x32_bf16(false, a, false, b0,
                                                   (short)0, acc0, false, false);
    acc1 = __builtin_amdgcn_wmma_f32_16x16x32_bf16(false, a, false, b1,
                                                   (short)0, acc1, false, false);

    if (kt + 1 < nk) lstore(cur ^ 1);
    __syncthreads();
  }

  const int col0 = n0 + cw * 32 + m;
  const int col1 = col0 + 16;
  const float bv0 = bias[col0], bv1 = bias[col1];
#pragma unroll
  for (int v = 0; v < 8; ++v) {
    int row = m0 + rw * 16 + v + half * 8;
    float y0 = acc0[v] + bv0;
    float y1 = acc1[v] + bv1;
    if (act == 1) { y0 = gelu_exact(y0); y1 = gelu_exact(y1); }
    Y[(size_t)row * N + col0] = y0;
    Y[(size_t)row * N + col1] = y1;
  }
}

// ---------------------------------------------------------------------------
// Attention for one (sequence, head): O = softmax(Q K^T / sqrt(E)) V, E = 32.
// LP/SP: padded L/S (LP mult of 16, SP mult of 32). grid=(nseq, H),
// block = LP*2 threads (LP/16 waves). Head h occupies cols [h*32, h*32+32).
// Qs/Ks: [row][e] (contiguous-k frags for Q@K^T); VsT: [e][s] (contiguous-k
// frags for P@V).
// ---------------------------------------------------------------------------
template <int LP, int SP>
__global__ __launch_bounds__(LP * 2) void attn_kernel(
    const float* __restrict__ Q, const float* __restrict__ K,
    const float* __restrict__ V, float* __restrict__ O,
    int L, int S, float scale) {
  constexpr int SCP = SP + 8;
  __shared__ __bf16 Qs[LP][40];
  __shared__ __bf16 Ks[SP][40];
  __shared__ __bf16 VsT[32][SCP];
  __shared__ float  Sc[LP][SCP];
  __shared__ __bf16 Ps[LP][SCP];

  const int tid  = threadIdx.x;
  const int wv   = tid >> 5, lane = tid & 31;
  const int m    = lane & 15, half = lane >> 4;
  const int T    = LP * 2;
  const int seq  = blockIdx.x, head = blockIdx.y;

  const float* qb = Q + (size_t)seq * L * 256 + head * 32;
  const float* kb = K + (size_t)seq * S * 256 + head * 32;
  const float* vb = V + (size_t)seq * S * 256 + head * 32;

  for (int idx = tid; idx < LP * 32; idx += T) {
    int r = idx >> 5, c = idx & 31;
    Qs[r][c] = (r < L) ? (__bf16)(qb[(size_t)r * 256 + c] * scale) : (__bf16)0.0f;
  }
  for (int idx = tid; idx < SP * 32; idx += T) {
    int r = idx >> 5, c = idx & 31;
    Ks[r][c]  = (r < S) ? (__bf16)kb[(size_t)r * 256 + c] : (__bf16)0.0f;
    VsT[c][r] = (r < S) ? (__bf16)vb[(size_t)r * 256 + c] : (__bf16)0.0f;
  }
  __syncthreads();

  // scores = (Q*scale) @ K^T   (K dim = E = 32 -> exactly one WMMA per tile)
  {
    v16bf a;
    const __bf16* qrow = &Qs[wv * 16 + m][0];
#pragma unroll
    for (int e = 0; e < 16; ++e) a[e] = qrow[kmap(e, half)];
#pragma unroll
    for (int ct = 0; ct < SP / 16; ++ct) {
      v16bf b;
      const __bf16* krow = &Ks[ct * 16 + m][0];
#pragma unroll
      for (int e = 0; e < 16; ++e) b[e] = krow[kmap(e, half)];
      v8f acc = {0.f,0.f,0.f,0.f,0.f,0.f,0.f,0.f};
      acc = __builtin_amdgcn_wmma_f32_16x16x32_bf16(false, a, false, b,
                                                    (short)0, acc, false, false);
#pragma unroll
      for (int v = 0; v < 8; ++v)
        Sc[wv * 16 + v + half * 8][ct * 16 + m] = acc[v];
    }
  }
  __syncthreads();

  // row softmax (pad cols/rows zeroed so they contribute nothing to P@V)
  if (tid < LP) {
    int r = tid;
    if (r < L) {
      float mx = -3.0e38f;
      for (int s = 0; s < S; ++s) mx = fmaxf(mx, Sc[r][s]);
      float sum = 0.f;
      for (int s = 0; s < S; ++s) { float e = expf(Sc[r][s] - mx); Sc[r][s] = e; sum += e; }
      float inv = 1.0f / sum;
      for (int s = 0; s < SP; ++s)
        Ps[r][s] = (s < S) ? (__bf16)(Sc[r][s] * inv) : (__bf16)0.0f;
    } else {
      for (int s = 0; s < SP; ++s) Ps[r][s] = (__bf16)0.0f;
    }
  }
  __syncthreads();

  // O = P @ V (K dim = SP, E split into two 16-col tiles)
  {
    v8f acc0 = {0.f,0.f,0.f,0.f,0.f,0.f,0.f,0.f};
    v8f acc1 = acc0;
#pragma unroll
    for (int kt = 0; kt < SP / 32; ++kt) {
      const __bf16* prow  = &Ps[wv * 16 + m][kt * 32];
      const __bf16* vrow0 = &VsT[m][kt * 32];
      const __bf16* vrow1 = &VsT[16 + m][kt * 32];
      v16bf a, b0, b1;
#pragma unroll
      for (int e = 0; e < 16; ++e) {
        const int kk = kmap(e, half);
        a[e]  = prow[kk];
        b0[e] = vrow0[kk];
        b1[e] = vrow1[kk];
      }
      acc0 = __builtin_amdgcn_wmma_f32_16x16x32_bf16(false, a, false, b0,
                                                     (short)0, acc0, false, false);
      acc1 = __builtin_amdgcn_wmma_f32_16x16x32_bf16(false, a, false, b1,
                                                     (short)0, acc1, false, false);
    }
    float* ob = O + (size_t)seq * L * 256 + head * 32;
#pragma unroll
    for (int v = 0; v < 8; ++v) {
      int l = wv * 16 + v + half * 8;
      if (l < L) {
        ob[(size_t)l * 256 + m]      = acc0[v];
        ob[(size_t)l * 256 + 16 + m] = acc1[v];
      }
    }
  }
}

// ---------------------------------------------------------------------------
// Residual add + LayerNorm over last dim 256 (eps = 1e-3). Wave per row,
// 8 rows per 256-thread block. wave32 shuffle reductions.
// ---------------------------------------------------------------------------
__global__ __launch_bounds__(256) void add_ln_kernel(
    const float* __restrict__ A, const float* __restrict__ Bv,
    const float* __restrict__ g, const float* __restrict__ beta,
    float* __restrict__ Y) {
  const int tid  = threadIdx.x;
  const int wv   = tid >> 5, lane = tid & 31;
  const size_t row = (size_t)blockIdx.x * 8 + wv;
  const float* a = A  + row * 256;
  const float* b = Bv + row * 256;

  float v[8];
  float s = 0.f;
#pragma unroll
  for (int i = 0; i < 8; ++i) { v[i] = a[lane + i * 32] + b[lane + i * 32]; s += v[i]; }
#pragma unroll
  for (int off = 16; off > 0; off >>= 1) s += __shfl_xor(s, off, 32);
  const float mean = s * (1.0f / 256.0f);

  float q = 0.f;
#pragma unroll
  for (int i = 0; i < 8; ++i) { float d = v[i] - mean; q += d * d; }
#pragma unroll
  for (int off = 16; off > 0; off >>= 1) q += __shfl_xor(q, off, 32);
  const float rstd = rsqrtf(q * (1.0f / 256.0f) + 1e-3f);

  float* y = Y + row * 256;
#pragma unroll
  for (int i = 0; i < 8; ++i) {
    int c = lane + i * 32;
    y[c] = (v[i] - mean) * rstd * g[c] + beta[c];
  }
}

// ---------------------------------------------------------------------------
// Layout shuffles (float4 granularity; D=256 -> 64 float4 per row)
// ---------------------------------------------------------------------------
// (b*ts_d, seg, d) -> (b*seg, ts_d, d)
__global__ __launch_bounds__(256) void permute_send_kernel(
    const float* __restrict__ in, float* __restrict__ out) {
  size_t idx = (size_t)blockIdx.x * 256 + threadIdx.x;
  int c4 = (int)(idx & 63);
  int ro = (int)(idx >> 6);
  int t = ro & 31, bs = ro >> 5;
  int s = bs & 63, b = bs >> 6;
  int ri = (b * 32 + t) * 64 + s;
  ((float4*)out)[(size_t)ro * 64 + c4] = ((const float4*)in)[(size_t)ri * 64 + c4];
}

// (b*seg, ts_d, d) -> (b, ts_d, seg, d)
__global__ __launch_bounds__(256) void permute_out_kernel(
    const float* __restrict__ in, float* __restrict__ out) {
  size_t idx = (size_t)blockIdx.x * 256 + threadIdx.x;
  int c4 = (int)(idx & 63);
  int ro = (int)(idx >> 6);
  int s = ro & 63, bt = ro >> 6;
  int t = bt & 31, b = bt >> 5;
  int ri = (b * 64 + s) * 32 + t;
  ((float4*)out)[(size_t)ro * 64 + c4] = ((const float4*)in)[(size_t)ri * 64 + c4];
}

// batch_router[j, f, :] = router[j / B, f, :]  (seg-major tf.repeat, B=16)
__global__ __launch_bounds__(256) void router_repeat_kernel(
    const float* __restrict__ router, float* __restrict__ out) {
  size_t idx = (size_t)blockIdx.x * 256 + threadIdx.x;  // over 10240*64 float4
  int c4 = (int)(idx & 63);
  int orow = (int)(idx >> 6);          // j * 10 + f
  int f = orow % 10;
  int j = orow / 10;
  int irow = (j >> 4) * 10 + f;        // (j / B) * FACTOR + f
  ((float4*)out)[(size_t)orow * 64 + c4] =
      ((const float4*)router)[(size_t)irow * 64 + c4];
}

// ---------------------------------------------------------------------------
// Host-side orchestration
// ---------------------------------------------------------------------------
extern "C" void kernel_launch(void* const* d_in, const int* in_sizes, int n_in,
                              void* d_out, int out_size, void* d_ws, size_t ws_size,
                              hipStream_t stream) {
  (void)in_sizes; (void)n_in; (void)out_size;

  const float* x      = (const float*)d_in[0];
  const float* router = (const float*)d_in[1];
  const float *tq_w=(const float*)d_in[2],  *tq_b=(const float*)d_in[3];
  const float *tk_w=(const float*)d_in[4],  *tk_b=(const float*)d_in[5];
  const float *tv_w=(const float*)d_in[6],  *tv_b=(const float*)d_in[7];
  const float *to_w=(const float*)d_in[8],  *to_b=(const float*)d_in[9];
  const float *sq_w=(const float*)d_in[10], *sq_b=(const float*)d_in[11];
  const float *sk_w=(const float*)d_in[12], *sk_b=(const float*)d_in[13];
  const float *sv_w=(const float*)d_in[14], *sv_b=(const float*)d_in[15];
  const float *so_w=(const float*)d_in[16], *so_b=(const float*)d_in[17];
  const float *rq_w=(const float*)d_in[18], *rq_b=(const float*)d_in[19];
  const float *rk_w=(const float*)d_in[20], *rk_b=(const float*)d_in[21];
  const float *rv_w=(const float*)d_in[22], *rv_b=(const float*)d_in[23];
  const float *ro_w=(const float*)d_in[24], *ro_b=(const float*)d_in[25];
  const float *n1_g=(const float*)d_in[26], *n1_b=(const float*)d_in[27];
  const float *n2_g=(const float*)d_in[28], *n2_b=(const float*)d_in[29];
  const float *n3_g=(const float*)d_in[30], *n3_b=(const float*)d_in[31];
  const float *n4_g=(const float*)d_in[32], *n4_b=(const float*)d_in[33];
  const float *m1w1=(const float*)d_in[34], *m1b1=(const float*)d_in[35];
  const float *m1w2=(const float*)d_in[36], *m1b2=(const float*)d_in[37];
  const float *m2w1=(const float*)d_in[38], *m2b1=(const float*)d_in[39];
  const float *m2w2=(const float*)d_in[40], *m2b2=(const float*)d_in[41];

  const size_t NT = 32768;   // B*TS_D*SEG tokens (also B*SEG*TS_D)
  const size_t NR = 10240;   // B*SEG*FACTOR router tokens
  const int Dm = 256, DF = 1024;
  const float scale = 0.17677669529663687f;  // 1/sqrt(32)

  float* ws = (float*)d_ws;
  float* Wq  = ws;                 // NT*D
  float* Wk  = Wq  + NT * Dm;      // NT*D
  float* Wv  = Wk  + NT * Dm;      // NT*D
  float* Wo  = Wv  + NT * Dm;      // NT*D
  float* Wt  = Wo  + NT * Dm;      // NT*D
  float* Wf  = Wt  + NT * Dm;      // NT*D
  float* Wg  = Wf  + NT * Dm;      // NT*D  (dim_send)
  float* Wh  = Wg  + NT * Dm;      // NT*DFF
  float* Rr  = Wh  + NT * DF;      // NR*D  (batch_router)
  float* Rq  = Rr  + NR * Dm;      // NR*D
  float* Rob = Rq  + NR * Dm;      // NR*D  (sender attn out)
  float* Rdb = Rob + NR * Dm;      // NR*D  (dim_buffer)
  float* Rk2 = Rdb + NR * Dm;      // NR*D
  float* Rv2 = Rk2 + NR * Dm;      // NR*D
  const size_t need = (size_t)(Rv2 + NR * Dm - ws) * sizeof(float);
  if (ws_size < need) return;

  const dim3 gBig((unsigned)(NT / 64), Dm / 64);   // (512, 4)
  const dim3 gBigF((unsigned)(NT / 64), DF / 64);  // (512, 16)
  const dim3 gRtr((unsigned)(NR / 64), Dm / 64);   // (160, 4)
  const unsigned copyBlocks = (unsigned)(NT * 64 / 256);   // float4 copies
  const unsigned rtrBlocks  = (unsigned)(NR * 64 / 256);

  // ---------------- Stage 1: time attention (512 seqs of L=S=64) -----------
  gemm_bias_act<<<gBig, 256, 0, stream>>>(x, tq_w, tq_b, Wq, Dm, Dm, 0);
  gemm_bias_act<<<gBig, 256, 0, stream>>>(x, tk_w, tk_b, Wk, Dm, Dm, 0);
  gemm_bias_act<<<gBig, 256, 0, stream>>>(x, tv_w, tv_b, Wv, Dm, Dm, 0);
  attn_kernel<64, 64><<<dim3(512, 8), 128, 0, stream>>>(Wq, Wk, Wv, Wo, 64, 64, scale);
  gemm_bias_act<<<gBig, 256, 0, stream>>>(Wo, to_w, to_b, Wt, Dm, Dm, 0);
  add_ln_kernel<<<(unsigned)(NT / 8), 256, 0, stream>>>(x, Wt, n1_g, n1_b, Wf);
  gemm_bias_act<<<gBigF, 256, 0, stream>>>(Wf, m1w1, m1b1, Wh, Dm, DF, 1);
  gemm_bias_act<<<gBig, 256, 0, stream>>>(Wh, m1w2, m1b2, Wt, DF, Dm, 0);
  add_ln_kernel<<<(unsigned)(NT / 8), 256, 0, stream>>>(Wf, Wt, n2_g, n2_b, Wq);

  // ---------------- Stage 2: routed cross-dimension attention --------------
  permute_send_kernel<<<copyBlocks, 256, 0, stream>>>(Wq, Wg);
  router_repeat_kernel<<<rtrBlocks, 256, 0, stream>>>(router, Rr);
  // sender: Q = batch_router (L=10), K/V = dim_send (S=32)
  gemm_bias_act<<<gRtr, 256, 0, stream>>>(Rr, sq_w, sq_b, Rq, Dm, Dm, 0);
  gemm_bias_act<<<gBig, 256, 0, stream>>>(Wg, sk_w, sk_b, Wk, Dm, Dm, 0);
  gemm_bias_act<<<gBig, 256, 0, stream>>>(Wg, sv_w, sv_b, Wv, Dm, Dm, 0);
  attn_kernel<16, 32><<<dim3(1024, 8), 32, 0, stream>>>(Rq, Wk, Wv, Rob, 10, 32, scale);
  gemm_bias_act<<<gRtr, 256, 0, stream>>>(Rob, so_w, so_b, Rdb, Dm, Dm, 0);
  // receiver: Q = dim_send (L=32), K/V = dim_buffer (S=10)
  gemm_bias_act<<<gBig, 256, 0, stream>>>(Wg, rq_w, rq_b, Wq, Dm, Dm, 0);
  gemm_bias_act<<<gRtr, 256, 0, stream>>>(Rdb, rk_w, rk_b, Rk2, Dm, Dm, 0);
  gemm_bias_act<<<gRtr, 256, 0, stream>>>(Rdb, rv_w, rv_b, Rv2, Dm, Dm, 0);
  attn_kernel<32, 32><<<dim3(1024, 8), 64, 0, stream>>>(Wq, Rk2, Rv2, Wo, 32, 10, scale);
  gemm_bias_act<<<gBig, 256, 0, stream>>>(Wo, ro_w, ro_b, Wt, Dm, Dm, 0);
  add_ln_kernel<<<(unsigned)(NT / 8), 256, 0, stream>>>(Wg, Wt, n3_g, n3_b, Wf);
  gemm_bias_act<<<gBigF, 256, 0, stream>>>(Wf, m2w1, m2b1, Wh, Dm, DF, 1);
  gemm_bias_act<<<gBig, 256, 0, stream>>>(Wh, m2w2, m2b2, Wt, DF, Dm, 0);
  add_ln_kernel<<<(unsigned)(NT / 8), 256, 0, stream>>>(Wf, Wt, n4_g, n4_b, Wq);

  permute_out_kernel<<<copyBlocks, 256, 0, stream>>>(Wq, (float*)d_out);
}
